// GAT_68143951118644
// MI455X (gfx1250) — matrix operand order repaired and checked by reading, not compiled, not run on hardware
//
#include <hip/hip_runtime.h>
#include <hip/hip_bf16.h>

typedef __attribute__((ext_vector_type(16))) _Float16 v16h;
typedef __attribute__((ext_vector_type(8)))  float    v8f;

#define NEG_SLOPE 0.2f
#define GAT_EPS 1e-16f

// ---------------------------------------------------------------- utilities

__global__ void k_f32_to_f16(const float* __restrict__ in, _Float16* __restrict__ out, int n) {
    int i = blockIdx.x * blockDim.x + threadIdx.x;
    if (i < n) out[i] = (_Float16)in[i];
}

__global__ void k_fill_u32(unsigned int* __restrict__ p, unsigned int bits, int n) {
    int i = blockIdx.x * blockDim.x + threadIdx.x;
    if (i < n) p[i] = bits;
}

// Pack f32 B[K,Nc] into f16 WMMA B-operand lane layout:
//   Bp[(((tn*(K/32) + kq)*32 + lane)*16 + i)] = B[(kq*32 + (lane>>4)*16 + i)*Nc + tn*16 + (lane&15)]
// so each (wave-lane, k-step, n-tile) reads one contiguous 32-byte chunk.
__global__ void k_pack_b(const float* __restrict__ B, _Float16* __restrict__ Bp, int K, int Nc) {
    int idx = blockIdx.x * blockDim.x + threadIdx.x;
    int KQ = K >> 5;
    int total = Nc * K;
    if (idx >= total) return;
    int i    = idx & 15;
    int lane = (idx >> 4) & 31;
    int kq   = (idx >> 9) % KQ;
    int tn   = (idx >> 9) / KQ;
    int k    = kq * 32 + (lane >> 4) * 16 + i;
    int n    = tn * 16 + (lane & 15);
    Bp[idx] = (_Float16)B[(size_t)k * Nc + n];
}

// ---------------------------------------------------------------- WMMA GEMM
// C[M,Nc] = A[M,K] x B[K,Nc]. One wave owns a 16-row stripe and all TN=Nc/16
// column tiles: A loaded once per k-step, reused across TN WMMAs.
// K compile-time -> fully unrolled -> 4*TN v_wmma per kernel body.
// Layouts per CDNA5 ISA 7.12.2 (A 16x32 f16, B 32x16 f16, D 16x16 f32).
template<int TN, int KK>
__global__ void k_gemm_wmma(const _Float16* __restrict__ A, const _Float16* __restrict__ Bp,
                            float* __restrict__ C, int tiles_m) {
    int wave = blockIdx.x * (blockDim.x >> 5) + (threadIdx.x >> 5);
    if (wave >= tiles_m) return;            // wave-uniform: EXEC stays all-ones
    int lane = threadIdx.x & 31;
    int l16  = lane & 15;
    int g    = lane >> 4;
    const int KQ = KK / 32;
    const int Nc = TN * 16;

    v8f acc[TN];
    #pragma unroll
    for (int t = 0; t < TN; ++t) acc[t] = (v8f){};

    const _Float16* Ab = A + (size_t)(wave * 16 + l16) * KK;
    #pragma unroll
    for (int kq = 0; kq < KQ; ++kq) {
        v16h a;
        #pragma unroll
        for (int j = 0; j < 8; ++j) {       // two contiguous 16B chunks -> b128 loads
            a[j]     = Ab[kq * 32 + g * 16 + j];
            a[8 + j] = Ab[kq * 32 + g * 16 + 8 + j];
        }
        #pragma unroll
        for (int tn = 0; tn < TN; ++tn) {
            const _Float16* bp = Bp + ((size_t)(tn * KQ + kq) * 32 + lane) * 16;
            v16h b;
            #pragma unroll
            for (int j = 0; j < 16; ++j) b[j] = bp[j];   // one contiguous 32B chunk
            acc[tn] = __builtin_amdgcn_wmma_f32_16x16x32_f16(
                false, a, false, b, (short)0, acc[tn], false, false);
        }
    }

    int row0 = wave * 16 + g * 8;
    #pragma unroll
    for (int tn = 0; tn < TN; ++tn) {
        int col = tn * 16 + l16;
        #pragma unroll
        for (int r = 0; r < 8; ++r)
            C[(size_t)(row0 + r) * Nc + col] = acc[tn][r];
    }
}

// ---------------------------------------------------------------- attention

// als[n,h] = <h[n,h,:], a_src[h,:]> ; ald likewise. One thread per (n,h).
__global__ void k_compute_al(const float* __restrict__ h, const float* __restrict__ a_src,
                             const float* __restrict__ a_dst, float* __restrict__ als,
                             float* __restrict__ ald, int N, int H, int C) {
    int i = blockIdx.x * blockDim.x + threadIdx.x;
    if (i >= N * H) return;
    int n = i / H, hd = i % H;
    const float* hp = h + (size_t)n * H * C + (size_t)hd * C;
    const float* as = a_src + hd * C;
    const float* ad = a_dst + hd * C;
    float s = 0.f, d = 0.f;
    for (int c = 0; c < C; ++c) { float v = hp[c]; s += v * as[c]; d += v * ad[c]; }
    als[i] = s; ald[i] = d;
}

__device__ __forceinline__ void edge_sd(const int* __restrict__ ei, int e, int E,
                                        int& s, int& d) {
    if (e < E) { s = ei[e]; d = ei[E + e]; }
    else       { s = e - E; d = e - E; }      // self loops appended
}

// float atomic max via sign-split (memory initialized to -inf bits 0xFF800000)
__device__ __forceinline__ void atomicMaxF(float* addr, float val) {
    if (val >= 0.f) atomicMax((int*)addr, __float_as_int(val));
    else            atomicMin((unsigned int*)addr, __float_as_uint(val));
}

__global__ void k_edge_max(const int* __restrict__ ei, int E, int Et,
                           const float* __restrict__ als, const float* __restrict__ ald,
                           float* __restrict__ m, int H) {
    int t = blockIdx.x * blockDim.x + threadIdx.x;
    if (t >= Et * H) return;
    int e = t / H, hd = t % H;
    int s, d; edge_sd(ei, e, E, s, d);
    float v = als[s * H + hd] + ald[d * H + hd];
    v = v > 0.f ? v : NEG_SLOPE * v;
    atomicMaxF(&m[d * H + hd], v);
}

__global__ void k_edge_exp(const int* __restrict__ ei, int E, int Et,
                           const float* __restrict__ als, const float* __restrict__ ald,
                           const float* __restrict__ m, float* __restrict__ ex,
                           float* __restrict__ ssum, int H) {
    int t = blockIdx.x * blockDim.x + threadIdx.x;
    if (t >= Et * H) return;
    int e = t / H, hd = t % H;
    int s, d; edge_sd(ei, e, E, s, d);
    float v = als[s * H + hd] + ald[d * H + hd];
    v = v > 0.f ? v : NEG_SLOPE * v;
    float exv = __expf(v - m[d * H + hd]);
    ex[t] = exv;
    atomicAdd(&ssum[d * H + hd], exv);
}

// One thread per (edge, 4 features): float4 gather, 4x GLOBAL_ATOMIC_ADD_F32 scatter.
// Requires C % 4 == 0 so the quad stays within one head.
__global__ void k_edge_agg4(const int* __restrict__ ei, int E, int Et,
                            const float* __restrict__ ex, const float* __restrict__ ssum,
                            const float* __restrict__ hval, float* __restrict__ agg,
                            int H, int C) {
    int F = H * C;
    int Q = F >> 2;
    int t = blockIdx.x * blockDim.x + threadIdx.x;
    if (t >= Et * Q) return;
    int e = t / Q, q = t % Q;
    int f0 = q << 2;
    int hd = f0 / C;
    int s, d; edge_sd(ei, e, E, s, d);
    float alpha = ex[e * H + hd] / (ssum[d * H + hd] + GAT_EPS);
    const float4 hv = *(const float4*)(hval + (size_t)s * F + f0);
    float* ap = agg + (size_t)d * F + f0;
    atomicAdd(ap + 0, alpha * hv.x);
    atomicAdd(ap + 1, alpha * hv.y);
    atomicAdd(ap + 2, alpha * hv.z);
    atomicAdd(ap + 3, alpha * hv.w);
}

// ELU(agg + b) -> f16 (feeds layer-2 WMMA GEMM)
__global__ void k_bias_elu_f16(const float* __restrict__ agg, const float* __restrict__ b,
                               _Float16* __restrict__ out16, int total, int F) {
    int i = blockIdx.x * blockDim.x + threadIdx.x;
    if (i >= total) return;
    float v = agg[i] + b[i % F];
    v = v > 0.f ? v : (__expf(v) - 1.f);
    out16[i] = (_Float16)v;
}

__global__ void k_add_bias(float* __restrict__ out, const float* __restrict__ b,
                           int total, int F) {
    int i = blockIdx.x * blockDim.x + threadIdx.x;
    if (i < total) out[i] += b[i % F];
}

// ---------------------------------------------------------------- driver

extern "C" void kernel_launch(void* const* d_in, const int* in_sizes, int n_in,
                              void* d_out, int out_size, void* d_ws, size_t ws_size,
                              hipStream_t stream) {
    const float* x   = (const float*)d_in[0];
    const int*   ei  = (const int*)  d_in[1];
    const float* W1  = (const float*)d_in[2];
    const float* a1s = (const float*)d_in[3];
    const float* a1d = (const float*)d_in[4];
    const float* b1  = (const float*)d_in[5];
    const float* W2  = (const float*)d_in[6];
    const float* a2s = (const float*)d_in[7];
    const float* a2d = (const float*)d_in[8];
    const float* b2  = (const float*)d_in[9];

    const int IN = 128, HID = 128, OUT = 64, H1 = 4, C1 = HID / H1;
    const int N  = in_sizes[0] / IN;
    const int E  = in_sizes[1] / 2;
    const int Et = E + N;                      // with self loops

    // workspace carve-out
    char* ws = (char*)d_ws;
    size_t off = 0;
    auto carve = [&](size_t bytes) -> void* {
        void* p = ws + off;
        off += bytes;
        off = (off + 255) & ~(size_t)255;
        return p;
    };
    _Float16* xh   = (_Float16*)carve((size_t)N * IN * 2);
    _Float16* W1p  = (_Float16*)carve((size_t)IN * HID * 2);   // packed B layout
    _Float16* W2p  = (_Float16*)carve((size_t)HID * OUT * 2);  // packed B layout
    float*    h1   = (float*)   carve((size_t)N * HID * 4);
    float*    agg1 = (float*)   carve((size_t)N * HID * 4);
    _Float16* h1e  = (_Float16*)carve((size_t)N * HID * 2);
    float*    h2   = (float*)   carve((size_t)N * OUT * 4);
    float*    als1 = (float*)   carve((size_t)N * H1 * 4);
    float*    ald1 = (float*)   carve((size_t)N * H1 * 4);
    float*    m1   = (float*)   carve((size_t)N * H1 * 4);
    float*    s1   = (float*)   carve((size_t)N * H1 * 4);
    float*    ex1  = (float*)   carve((size_t)Et * H1 * 4);
    float*    als2 = (float*)   carve((size_t)N * 4);
    float*    ald2 = (float*)   carve((size_t)N * 4);
    float*    m2   = (float*)   carve((size_t)N * 4);
    float*    s2   = (float*)   carve((size_t)N * 4);
    float*    ex2  = (float*)   carve((size_t)Et * 4);
    float*    out  = (float*)d_out;

    const int T = 256;
    auto cdiv = [](long long a, long long b) { return (int)((a + b - 1) / b); };

    // operand prep
    k_f32_to_f16<<<cdiv((long long)N * IN, T), T, 0, stream>>>(x, xh, N * IN);
    k_pack_b<<<cdiv(IN * HID, T), T, 0, stream>>>(W1, W1p, IN, HID);
    k_pack_b<<<cdiv(HID * OUT, T), T, 0, stream>>>(W2, W2p, HID, OUT);

    // ---------------- layer 1 ----------------
    {   // h1 = x @ W1  (N x 128 x 128): wave owns 16 rows x all 8 col tiles
        int tiles_m = N / 16;
        k_gemm_wmma<8, 128><<<cdiv(tiles_m, 8), 256, 0, stream>>>(xh, W1p, h1, tiles_m);
    }
    k_compute_al<<<cdiv((long long)N * H1, T), T, 0, stream>>>(h1, a1s, a1d, als1, ald1, N, H1, C1);
    k_fill_u32<<<cdiv((long long)N * H1, T), T, 0, stream>>>((unsigned int*)m1, 0xFF800000u, N * H1);
    k_fill_u32<<<cdiv((long long)N * H1, T), T, 0, stream>>>((unsigned int*)s1, 0u, N * H1);
    k_fill_u32<<<cdiv((long long)N * HID, T), T, 0, stream>>>((unsigned int*)agg1, 0u, N * HID);
    k_edge_max<<<cdiv((long long)Et * H1, T), T, 0, stream>>>(ei, E, Et, als1, ald1, m1, H1);
    k_edge_exp<<<cdiv((long long)Et * H1, T), T, 0, stream>>>(ei, E, Et, als1, ald1, m1, ex1, s1, H1);
    k_edge_agg4<<<cdiv((long long)Et * (HID / 4), T), T, 0, stream>>>(ei, E, Et, ex1, s1, h1, agg1, H1, C1);
    k_bias_elu_f16<<<cdiv((long long)N * HID, T), T, 0, stream>>>(agg1, b1, h1e, N * HID, HID);

    // ---------------- layer 2 ----------------
    {   // h2 = elu(h1_out) @ W2  (N x 128 x 64): wave owns 16 rows x all 4 col tiles
        int tiles_m = N / 16;
        k_gemm_wmma<4, 128><<<cdiv(tiles_m, 8), 256, 0, stream>>>(h1e, W2p, h2, tiles_m);
    }
    k_compute_al<<<cdiv(N, T), T, 0, stream>>>(h2, a2s, a2d, als2, ald2, N, 1, OUT);
    k_fill_u32<<<cdiv(N, T), T, 0, stream>>>((unsigned int*)m2, 0xFF800000u, N);
    k_fill_u32<<<cdiv(N, T), T, 0, stream>>>((unsigned int*)s2, 0u, N);
    k_fill_u32<<<cdiv((long long)N * OUT, T), T, 0, stream>>>((unsigned int*)out, 0u, N * OUT);
    k_edge_max<<<cdiv((long long)Et, T), T, 0, stream>>>(ei, E, Et, als2, ald2, m2, 1);
    k_edge_exp<<<cdiv((long long)Et, T), T, 0, stream>>>(ei, E, Et, als2, ald2, m2, ex2, s2, 1);
    k_edge_agg4<<<cdiv((long long)Et * (OUT / 4), T), T, 0, stream>>>(ei, E, Et, ex2, s2, h2, out, 1, OUT);
    k_add_bias<<<cdiv((long long)N * OUT, T), T, 0, stream>>>(out, b2, N * OUT, OUT);
}
